// Seq2Seq_86182813761868
// MI455X (gfx1250) — matrix operand order repaired and checked by reading, not compiled, not run on hardware
//
#include <hip/hip_runtime.h>

// ---------------------------------------------------------------------------
// Seq2Seq GRU (2-layer biGRU encoder + 4-layer GRU decoder + linear head)
// MI455X / gfx1250, wave32, V_WMMA_F32_16X16X32_F16 everywhere.
//   Phase A: parallel WMMA GEMMs for time-independent gate inputs (gi).
//   Phase B: persistent single-WGP recurrence kernels (512 thr = 16 waves),
//            h state: f16 in LDS for matmul A-frags, f32 master in VGPRs.
//            B-fragments shared across two batch tiles per K-step; Gi for the
//            next timestep prefetched (global_prefetch_b8) behind the WMMAs.
// ---------------------------------------------------------------------------

typedef __attribute__((ext_vector_type(16))) _Float16 v16h;
typedef __attribute__((ext_vector_type(8)))  float    v8f;

#define DEV __device__ __forceinline__

union FragU { v16h v; uint4 q[2]; };

// A-matrix 16x32 f16 fragment (ISA 7.12.2): lane r=lane&15 holds row M=r,
// halfs 0..7 = K in [8*half, 8*half+8), halfs 8..15 = K in [16+8*half, ...)
DEV v16h load_a_frag(const _Float16* p, int half) {
  FragU u;
  u.q[0] = *(const uint4*)(p + half * 8);
  u.q[1] = *(const uint4*)(p + 16 + half * 8);
  return u.v;
}
// B-matrix 32x16 f16 fragment: lane holds col N=lane&15, same K order as A-frag.
// We feed B[k][n] = W[n][k] (X @ W.T), so per-lane data is a contiguous W row.
DEV v16h load_b_frag(const _Float16* p, int half) {
  FragU u;
  u.q[0] = *(const uint4*)(p + half * 8);
  u.q[1] = *(const uint4*)(p + 16 + half * 8);
  return u.v;
}
DEV v8f wmma32(v16h a, v16h b, v8f c) {
  return __builtin_amdgcn_wmma_f32_16x16x32_f16(false, a, false, b, (short)0, c,
                                                false, false);
}
DEV float sigmoidf(float x) { return 1.0f / (1.0f + __expf(-x)); }

union Pk2 { unsigned u; _Float16 h[2]; };
DEV unsigned pack2h(float a, float b) {
  Pk2 x; x.h[0] = (_Float16)a; x.h[1] = (_Float16)b; return x.u;
}

// ---------------------------------------------------------------------------
__global__ void f32_to_f16_kernel(const float* __restrict__ s,
                                  _Float16* __restrict__ d, int n) {
  int i = blockIdx.x * blockDim.x + threadIdx.x;
  if (i < n) d[i] = (_Float16)s[i];
}

// out[m][0..767] = A[m][:] @ W.T + bias   (A: MxK f16, W: 768xK f16, out f32)
__global__ __launch_bounds__(256) void gi_gemm_kernel(
    const _Float16* __restrict__ A, int lda,
    const _Float16* __restrict__ W, int ldw,
    const float* __restrict__ bias, float* __restrict__ out, int M, int K) {
  const int wave = blockIdx.x * (blockDim.x >> 5) + (threadIdx.x >> 5);
  const int lane = threadIdx.x & 31, r = lane & 15, half = lane >> 4;
  const int tn = 48;                      // 768 / 16
  const int total = (M / 16) * tn;
  if (wave >= total) return;              // wave-uniform: EXEC stays all-ones
  const int mt = wave / tn, nt = wave % tn;
  const _Float16* arow = A + (long)(mt * 16 + r) * lda;
  const _Float16* wrow = W + (long)(nt * 16 + r) * ldw;
  v8f acc = {};
  for (int k0 = 0; k0 < K; k0 += 32)
    acc = wmma32(load_a_frag(arow + k0, half), load_b_frag(wrow + k0, half), acc);
  const float bj = bias[nt * 16 + r];
#pragma unroll
  for (int v = 0; v < 8; ++v) {
    const int m = mt * 16 + v + 8 * half;
    out[(long)m * 768 + nt * 16 + r] = acc[v] + bj;
  }
}

// ---------------------------------------------------------------------------
// Encoder recurrence: one persistent workgroup per direction (grid=2).
// 16 waves: wave w owns col-tile cc=w and all 4 batch tiles, processed as two
// joint pairs so each Whh B-fragment is reused by two A-fragments.
struct EncDir {
  const float* Gi;       // [rows][768], row = t*tS + b*bS  (includes bih)
  const _Float16* Whh;   // [768][256] f16
  const float* bhh;      // [768]
  _Float16* Y;           // [T*64][512] (f16, next layer's GEMM input) or null
  float* hT;             // [64][256] final hidden
  int reverse, tS, bS, yoff;
};

__global__ __launch_bounds__(512) void enc_rnn_kernel(EncDir df, EncDir db) {
  EncDir A = (blockIdx.x == 0) ? df : db;
  __shared__ __align__(32) _Float16 hbuf[2][64 * 256];   // 64 KB ping-pong
  const int lane = threadIdx.x & 31, w = threadIdx.x >> 5;   // 16 waves
  const int r = lane & 15, half = lane >> 4;
  const int cc = w;                                          // col tile 0..15
  for (int i = threadIdx.x; i < 64 * 256; i += blockDim.x)
    hbuf[0][i] = (_Float16)0.0f;
  float hreg[4][8];
#pragma unroll
  for (int bb = 0; bb < 4; ++bb)
#pragma unroll
    for (int v = 0; v < 8; ++v) hreg[bb][v] = 0.0f;
  const float bh0 = A.bhh[0 * 256 + cc * 16 + r];
  const float bh1 = A.bhh[1 * 256 + cc * 16 + r];
  const float bh2 = A.bhh[2 * 256 + cc * 16 + r];
  const _Float16* w0 = A.Whh + (long)(0 * 256 + cc * 16 + r) * 256;
  const _Float16* w1 = A.Whh + (long)(1 * 256 + cc * 16 + r) * 256;
  const _Float16* w2 = A.Whh + (long)(2 * 256 + cc * 16 + r) * 256;
  __syncthreads();
  for (int step = 0; step < 128; ++step) {
    const int t = A.reverse ? (127 - step) : step;
    const _Float16* hc = hbuf[step & 1];
    _Float16* hn = hbuf[(step & 1) ^ 1];
    // prefetch next timestep's Gi lines behind this step's WMMA work
    if (step + 1 < 128) {
      const int tnx = A.reverse ? (127 - (step + 1)) : (step + 1);
#pragma unroll
      for (int bb = 0; bb < 4; ++bb) {
        const int m = bb * 16 + r;   // lanes cover all 16 rows of the tile
        const long gb = ((long)tnx * A.tS + (long)m * A.bS) * 768 + cc * 16 + r;
        __builtin_prefetch(&A.Gi[gb], 0, 1);
        __builtin_prefetch(&A.Gi[gb + 256], 0, 1);
        __builtin_prefetch(&A.Gi[gb + 512], 0, 1);
      }
    }
#pragma unroll
    for (int g = 0; g < 2; ++g) {          // two joint pairs of batch tiles
      v8f A0[2] = {{}, {}}, A1[2] = {{}, {}}, A2[2] = {{}, {}};
      const _Float16* ar0 = hc + ((2 * g + 0) * 16 + r) * 256;
      const _Float16* ar1 = hc + ((2 * g + 1) * 16 + r) * 256;
#pragma unroll
      for (int k0 = 0; k0 < 256; k0 += 32) {
        const v16h b0 = load_b_frag(w0 + k0, half);
        const v16h b1 = load_b_frag(w1 + k0, half);
        const v16h b2 = load_b_frag(w2 + k0, half);
        const v16h f0 = load_a_frag(ar0 + k0, half);
        const v16h f1 = load_a_frag(ar1 + k0, half);
        A0[0] = wmma32(f0, b0, A0[0]);
        A1[0] = wmma32(f0, b1, A1[0]);
        A2[0] = wmma32(f0, b2, A2[0]);
        A0[1] = wmma32(f1, b0, A0[1]);
        A1[1] = wmma32(f1, b1, A1[1]);
        A2[1] = wmma32(f1, b2, A2[1]);
      }
#pragma unroll
      for (int p = 0; p < 2; ++p) {
        const int bb = 2 * g + p;
#pragma unroll
        for (int v = 0; v < 8; ++v) {
          const int m = bb * 16 + v + 8 * half;
          const long gib = ((long)t * A.tS + (long)m * A.bS) * 768 + cc * 16 + r;
          const float gr = A.Gi[gib];
          const float gz = A.Gi[gib + 256];
          const float gn = A.Gi[gib + 512];
          const float rr = sigmoidf(gr + A0[p][v] + bh0);
          const float zz = sigmoidf(gz + A1[p][v] + bh1);
          const float nn = tanhf(gn + rr * (A2[p][v] + bh2));
          const float hv = (1.0f - zz) * nn + zz * hreg[bb][v];
          hreg[bb][v] = hv;
          hn[m * 256 + cc * 16 + r] = (_Float16)hv;
          if (A.Y)
            A.Y[((long)t * 64 + m) * 512 + A.yoff + cc * 16 + r] = (_Float16)hv;
        }
      }
    }
    __syncthreads();
  }
#pragma unroll
  for (int bb = 0; bb < 4; ++bb)
#pragma unroll
    for (int v = 0; v < 8; ++v)
      A.hT[(bb * 16 + v + 8 * half) * 256 + cc * 16 + r] = hreg[bb][v];
}

// ---------------------------------------------------------------------------
// Decoder: single persistent workgroup (16 waves). Per timestep: 4 fused GRU
// layers (gi from in16, gh from h16; the n-gate's input/hidden halves kept in
// separate accumulators), then linear head feeding in16 for the next step.
struct DecLayerArg {
  const _Float16* Wih;   // [768][din]
  const _Float16* Whh;   // [768][256]
  const float* bih;
  const float* bhh;
  int din;               // 64 for layer 0, else 256
};

__global__ __launch_bounds__(512) void dec_rnn_kernel(
    DecLayerArg L0, DecLayerArg L1, DecLayerArg L2, DecLayerArg L3,
    const _Float16* __restrict__ linW, const float* __restrict__ linb,
    const float* __restrict__ hT, float* __restrict__ out) {
  __shared__ __align__(32) _Float16 h16[64 * 256];    // current layer h (f16)
  __shared__ __align__(32) _Float16 in16[64 * 256];   // layer input (f16)
  DecLayerArg Ls[4] = {L0, L1, L2, L3};
  const int lane = threadIdx.x & 31, w = threadIdx.x >> 5;   // 16 waves
  const int r = lane & 15, half = lane >> 4;
  const int cc = w;
  float hreg[4][4][8];
  float bs_r[4], bs_z[4], bi_n[4], bh_n[4];
#pragma unroll
  for (int l = 0; l < 4; ++l) {
    const int j0 = cc * 16 + r;
    bs_r[l] = Ls[l].bih[j0] + Ls[l].bhh[j0];
    bs_z[l] = Ls[l].bih[256 + j0] + Ls[l].bhh[256 + j0];
    bi_n[l] = Ls[l].bih[512 + j0];
    bh_n[l] = Ls[l].bhh[512 + j0];
#pragma unroll
    for (int bb = 0; bb < 4; ++bb)
#pragma unroll
      for (int v = 0; v < 8; ++v)
        hreg[l][bb][v] =
            hT[l * 64 * 256 + (bb * 16 + v + 8 * half) * 256 + j0];
  }
  for (int i = threadIdx.x; i < 64 * 256; i += blockDim.x)
    in16[i] = (_Float16)0.0f;                       // first decoder input = 0
  __syncthreads();

  for (int t = 0; t < 128; ++t) {
#pragma unroll
    for (int l = 0; l < 4; ++l) {
      // publish this layer's h (prev timestep) as f16 for the gh matmul
#pragma unroll
      for (int bb = 0; bb < 4; ++bb)
#pragma unroll
        for (int v = 0; v < 8; ++v)
          h16[(bb * 16 + v + 8 * half) * 256 + cc * 16 + r] =
              (_Float16)hreg[l][bb][v];
      __syncthreads();
      const int din = Ls[l].din;
      const _Float16* wi0 = Ls[l].Wih + (long)(0 * 256 + cc * 16 + r) * din;
      const _Float16* wi1 = Ls[l].Wih + (long)(1 * 256 + cc * 16 + r) * din;
      const _Float16* wi2 = Ls[l].Wih + (long)(2 * 256 + cc * 16 + r) * din;
      const _Float16* wh0 = Ls[l].Whh + (long)(0 * 256 + cc * 16 + r) * 256;
      const _Float16* wh1 = Ls[l].Whh + (long)(1 * 256 + cc * 16 + r) * 256;
      const _Float16* wh2 = Ls[l].Whh + (long)(2 * 256 + cc * 16 + r) * 256;
      unsigned stash[4][4];     // new h, packed f16x2 (v pairs), per batch tile
#pragma unroll
      for (int g = 0; g < 2; ++g) {        // two joint pairs of batch tiles
        v8f ar[2] = {{}, {}}, az[2] = {{}, {}};
        v8f ani[2] = {{}, {}}, anh[2] = {{}, {}};
        const _Float16* ir0 = in16 + ((2 * g + 0) * 16 + r) * 256;
        const _Float16* ir1 = in16 + ((2 * g + 1) * 16 + r) * 256;
        const _Float16* hr0 = h16 + ((2 * g + 0) * 16 + r) * 256;
        const _Float16* hr1 = h16 + ((2 * g + 1) * 16 + r) * 256;
        for (int k0 = 0; k0 < din; k0 += 32) {        // gi: input-side GEMM
          const v16h b0 = load_b_frag(wi0 + k0, half);
          const v16h b1 = load_b_frag(wi1 + k0, half);
          const v16h b2 = load_b_frag(wi2 + k0, half);
          const v16h f0 = load_a_frag(ir0 + k0, half);
          const v16h f1 = load_a_frag(ir1 + k0, half);
          ar[0] = wmma32(f0, b0, ar[0]);
          az[0] = wmma32(f0, b1, az[0]);
          ani[0] = wmma32(f0, b2, ani[0]);
          ar[1] = wmma32(f1, b0, ar[1]);
          az[1] = wmma32(f1, b1, az[1]);
          ani[1] = wmma32(f1, b2, ani[1]);
        }
#pragma unroll
        for (int k0 = 0; k0 < 256; k0 += 32) {        // gh: hidden-side GEMM
          const v16h b0 = load_b_frag(wh0 + k0, half);
          const v16h b1 = load_b_frag(wh1 + k0, half);
          const v16h b2 = load_b_frag(wh2 + k0, half);
          const v16h f0 = load_a_frag(hr0 + k0, half);
          const v16h f1 = load_a_frag(hr1 + k0, half);
          ar[0] = wmma32(f0, b0, ar[0]);
          az[0] = wmma32(f0, b1, az[0]);
          anh[0] = wmma32(f0, b2, anh[0]);
          ar[1] = wmma32(f1, b0, ar[1]);
          az[1] = wmma32(f1, b1, az[1]);
          anh[1] = wmma32(f1, b2, anh[1]);
        }
#pragma unroll
        for (int p = 0; p < 2; ++p) {
          const int bb = 2 * g + p;
#pragma unroll
          for (int k = 0; k < 4; ++k) {
            float hv2[2];
#pragma unroll
            for (int q = 0; q < 2; ++q) {
              const int v = 2 * k + q;
              const float rr = sigmoidf(ar[p][v] + bs_r[l]);
              const float zz = sigmoidf(az[p][v] + bs_z[l]);
              const float nn =
                  tanhf(ani[p][v] + bi_n[l] + rr * (anh[p][v] + bh_n[l]));
              const float hv = (1.0f - zz) * nn + zz * hreg[l][bb][v];
              hreg[l][bb][v] = hv;
              hv2[q] = hv;
            }
            stash[bb][k] = pack2h(hv2[0], hv2[1]);
          }
        }
      }
      __syncthreads();   // all in16/h16 reads done before overwriting in16
#pragma unroll
      for (int bb = 0; bb < 4; ++bb)
#pragma unroll
        for (int k = 0; k < 4; ++k) {
          Pk2 x; x.u = stash[bb][k];
          const int m0 = bb * 16 + 2 * k + 8 * half;
          in16[m0 * 256 + cc * 16 + r] = x.h[0];
          in16[(m0 + 1) * 256 + cc * 16 + r] = x.h[1];
        }
      __syncthreads();
    }
    // linear head: out_t = h3 @ linW.T + lin_b ; h3 (f16) is in in16
    const int lbb = w >> 2, lc2 = w & 3;     // 16 tiles over 16 waves
    v8f ao = {};
    {
      const _Float16* arow = in16 + (lbb * 16 + r) * 256;
      const _Float16* wrow = linW + (long)(lc2 * 16 + r) * 256;
#pragma unroll
      for (int k0 = 0; k0 < 256; k0 += 32)
        ao = wmma32(load_a_frag(arow + k0, half), load_b_frag(wrow + k0, half),
                    ao);
    }
    const float bl = linb[lc2 * 16 + r];
    float lst[8];
#pragma unroll
    for (int v = 0; v < 8; ++v) {
      const int m = lbb * 16 + v + 8 * half;
      lst[v] = ao[v] + bl;
      out[((long)m * 128 + t) * 64 + lc2 * 16 + r] = lst[v];  // [B][T][IN]
    }
    __syncthreads();     // lin reads of in16 complete
#pragma unroll
    for (int v = 0; v < 8; ++v)
      in16[(lbb * 16 + v + 8 * half) * 256 + lc2 * 16 + r] = (_Float16)lst[v];
    __syncthreads();
  }
}

// ---------------------------------------------------------------------------
extern "C" void kernel_launch(void* const* d_in, const int* in_sizes, int n_in,
                              void* d_out, int out_size, void* d_ws,
                              size_t ws_size, hipStream_t stream) {
  (void)in_sizes; (void)n_in; (void)out_size; (void)ws_size;
  const float* x = (const float*)d_in[0];
  const float *e_Wih[2][2], *e_Whh[2][2], *e_bih[2][2], *e_bhh[2][2];
  int idx = 1;
  for (int l = 0; l < 2; ++l)
    for (int d = 0; d < 2; ++d) {
      e_Wih[l][d] = (const float*)d_in[idx++];
      e_Whh[l][d] = (const float*)d_in[idx++];
      e_bih[l][d] = (const float*)d_in[idx++];
      e_bhh[l][d] = (const float*)d_in[idx++];
    }
  const float *dc_Wih[4], *dc_Whh[4], *dc_bih[4], *dc_bhh[4];
  for (int l = 0; l < 4; ++l) {
    dc_Wih[l] = (const float*)d_in[idx++];
    dc_Whh[l] = (const float*)d_in[idx++];
    dc_bih[l] = (const float*)d_in[idx++];
    dc_bhh[l] = (const float*)d_in[idx++];
  }
  const float* linW = (const float*)d_in[idx++];
  const float* linb = (const float*)d_in[idx++];

  // workspace allocator (~67 MB total)
  char* ws = (char*)d_ws;
  size_t off = 0;
  auto alloc = [&](size_t bytes) -> char* {
    off = (off + 255) & ~(size_t)255;
    char* p = ws + off;
    off += bytes;
    return p;
  };
  const int edin[2] = {64, 512};
  const int ddin[4] = {64, 256, 256, 256};
  _Float16* x16 = (_Float16*)alloc((size_t)64 * 128 * 64 * 2);
  _Float16 *eWih16[2][2], *eWhh16[2][2];
  for (int l = 0; l < 2; ++l)
    for (int d = 0; d < 2; ++d) {
      eWih16[l][d] = (_Float16*)alloc((size_t)768 * edin[l] * 2);
      eWhh16[l][d] = (_Float16*)alloc((size_t)768 * 256 * 2);
    }
  _Float16 *dWih16[4], *dWhh16[4];
  for (int l = 0; l < 4; ++l) {
    dWih16[l] = (_Float16*)alloc((size_t)768 * ddin[l] * 2);
    dWhh16[l] = (_Float16*)alloc((size_t)768 * 256 * 2);
  }
  _Float16* lin16 = (_Float16*)alloc((size_t)64 * 256 * 2);
  float* GiF = (float*)alloc((size_t)8192 * 768 * 4);   // reused enc0 -> enc1
  float* GiB = (float*)alloc((size_t)8192 * 768 * 4);
  _Float16* Y01 = (_Float16*)alloc((size_t)128 * 64 * 512 * 2);
  float* hT = (float*)alloc((size_t)4 * 64 * 256 * 4);

  auto cvt = [&](const float* s, _Float16* d, int n) {
    f32_to_f16_kernel<<<(n + 255) / 256, 256, 0, stream>>>(s, d, n);
  };
  cvt(x, x16, 64 * 128 * 64);
  for (int l = 0; l < 2; ++l)
    for (int d = 0; d < 2; ++d) {
      cvt(e_Wih[l][d], eWih16[l][d], 768 * edin[l]);
      cvt(e_Whh[l][d], eWhh16[l][d], 768 * 256);
    }
  for (int l = 0; l < 4; ++l) {
    cvt(dc_Wih[l], dWih16[l], 768 * ddin[l]);
    cvt(dc_Whh[l], dWhh16[l], 768 * 256);
  }
  cvt(linW, lin16, 64 * 256);

  const int tiles = (8192 / 16) * 48;        // 24576 tiles, 8 waves/block
  const int gblocks = (tiles + 7) / 8;

  // encoder layer 0: gi over all timesteps (rows = b*128 + t), then recurrence
  gi_gemm_kernel<<<gblocks, 256, 0, stream>>>(x16, 64, eWih16[0][0], 64,
                                              e_bih[0][0], GiF, 8192, 64);
  gi_gemm_kernel<<<gblocks, 256, 0, stream>>>(x16, 64, eWih16[0][1], 64,
                                              e_bih[0][1], GiB, 8192, 64);
  EncDir f0{GiF, eWhh16[0][0], e_bhh[0][0], Y01, hT + 0 * 16384, 0, 1, 128, 0};
  EncDir b0{GiB, eWhh16[0][1], e_bhh[0][1], Y01, hT + 1 * 16384, 1, 1, 128, 256};
  enc_rnn_kernel<<<2, 512, 0, stream>>>(f0, b0);

  // encoder layer 1: input = concat(yf,yb) in Y01 (rows = t*64 + b, K=512)
  gi_gemm_kernel<<<gblocks, 256, 0, stream>>>(Y01, 512, eWih16[1][0], 512,
                                              e_bih[1][0], GiF, 8192, 512);
  gi_gemm_kernel<<<gblocks, 256, 0, stream>>>(Y01, 512, eWih16[1][1], 512,
                                              e_bih[1][1], GiB, 8192, 512);
  EncDir f1{GiF, eWhh16[1][0], e_bhh[1][0], nullptr, hT + 2 * 16384, 0, 64, 1, 0};
  EncDir b1{GiB, eWhh16[1][1], e_bhh[1][1], nullptr, hT + 3 * 16384, 1, 64, 1, 0};
  enc_rnn_kernel<<<2, 512, 0, stream>>>(f1, b1);

  // decoder (fully sequential): one persistent workgroup
  DecLayerArg D0{dWih16[0], dWhh16[0], dc_bih[0], dc_bhh[0], 64};
  DecLayerArg D1{dWih16[1], dWhh16[1], dc_bih[1], dc_bhh[1], 256};
  DecLayerArg D2{dWih16[2], dWhh16[2], dc_bih[2], dc_bhh[2], 256};
  DecLayerArg D3{dWih16[3], dWhh16[3], dc_bih[3], dc_bhh[3], 256};
  dec_rnn_kernel<<<1, 512, 0, stream>>>(D0, D1, D2, D3, lin16, linb, hT,
                                        (float*)d_out);
}
